// _DTIHarmonic_44143673868857
// MI455X (gfx1250) — matrix operand order repaired and checked by reading, not compiled, not run on hardware
//
#include <hip/hip_runtime.h>
#include <math.h>

typedef __attribute__((ext_vector_type(16))) _Float16 v16h;
typedef __attribute__((ext_vector_type(8)))  _Float16 v8h;
typedef __attribute__((ext_vector_type(8)))  float    v8f;

#define BB 4
#define N1 64
#define N2 512
#define DD 128
#define LL 3
#define NI 8

// ---------------------------------------------------------------------------
// All-f16-operand GEMM, f32 accumulate. One wave = 16(M) x 64(N) tile:
// A fragment built once per k-step, reused by 4 v_wmma_f32_16x16x32_f16.
//   C[m,n] = sum_k A16[m,k] * B16[n,k]      (B supplied transposed, row-major NxK)
// Fragments are pure vector loads (16B/32B), zero in-loop conversion.
// Optional outputs: f32 C (+bias,+relu), f16 row-major C16, f16 transposed C16T.
// M mult of 16, N mult of 64, K mult of 32. Batched via grid.z (element strides).
// A/B k order contiguous (k = 16*(lane/16) + s, not the ISA 7.12.2 table); f32 C per ISA 7.12.2.
// ---------------------------------------------------------------------------
__global__ void k_gemm_wmma(const _Float16* __restrict__ A16, int lda, long sA,
                            const _Float16* __restrict__ B16, int ldbn, long sB,
                            float* __restrict__ C, int ldc, long sC,
                            _Float16* __restrict__ C16,
                            _Float16* __restrict__ C16T, int ldtc,
                            const float* __restrict__ bias, int relu,
                            int M, int N, int K) {
  const int lane = threadIdx.x & 31;
  const int tn0 = blockIdx.x * 4;       // four 16-wide N tiles per wave
  const int tm = blockIdx.y, bz = blockIdx.z;
  const _Float16* Ab = A16 + (long)bz * sA;
  const _Float16* Bb = B16 + (long)bz * sB;

  const int m  = lane & 15;             // A row / B-C column within tile
  const int kh = lane >> 4;             // lane half selects K sub-range
  const int row = tm * 16 + m;

  v8f c0 = {}, c1 = {}, c2 = {}, c3 = {};
  for (int k0 = 0; k0 < K; k0 += 32) {
    // A fragment: lane holds K = kh*16 .. kh*16+15 (same order as B)
    const _Float16* pa = Ab + (long)row * lda + k0;
    const v16h a = *(const v16h*)(pa + kh * 16);
    if (k0 + 32 < K) __builtin_prefetch(pa + 32, 0, 1);     // global_prefetch

    // 4 B fragments + 4 WMMAs sharing the A fragment
#pragma unroll
    for (int t = 0; t < 4; ++t) {
      const int col = (tn0 + t) * 16 + m;
      // lane holds K = kh*16 .. kh*16+15 of column 'col' -> one 32B load
      const v16h b = *(const v16h*)(Bb + (long)col * ldbn + k0 + kh * 16);
      if (t == 0)      c0 = __builtin_amdgcn_wmma_f32_16x16x32_f16(false, a, false, b, (short)0, c0, false, false);
      else if (t == 1) c1 = __builtin_amdgcn_wmma_f32_16x16x32_f16(false, a, false, b, (short)0, c1, false, false);
      else if (t == 2) c2 = __builtin_amdgcn_wmma_f32_16x16x32_f16(false, a, false, b, (short)0, c2, false, false);
      else             c3 = __builtin_amdgcn_wmma_f32_16x16x32_f16(false, a, false, b, (short)0, c3, false, false);
    }
  }
  // epilogue: lane half owns C rows 0-7 / 8-15 of the tile
  const int mbase = tm * 16 + kh * 8;
#pragma unroll
  for (int t = 0; t < 4; ++t) {
    const int col = (tn0 + t) * 16 + m;
    const v8f* cp = (t == 0) ? &c0 : (t == 1) ? &c1 : (t == 2) ? &c2 : &c3;
    const float bv = bias ? bias[col] : 0.0f;
#pragma unroll
    for (int r = 0; r < 8; ++r) {
      float v = (*cp)[r] + bv;
      if (relu) v = fmaxf(v, 0.0f);
      const int mr = mbase + r;
      if (C)    (C + (long)bz * sC)[(long)mr * ldc + col] = v;
      if (C16)  C16[(long)mr * ldc + col] = (_Float16)v;
      if (C16T) C16T[(long)col * ldtc + mr] = (_Float16)v;
    }
  }
}

// ---------------------------------------------------------------------------
// f32 -> f16 cast (weights pre-pass)
// ---------------------------------------------------------------------------
__global__ void k_cast(const float* __restrict__ src, _Float16* __restrict__ dst, int n) {
  int idx = blockIdx.x * blockDim.x + threadIdx.x;
  if (idx < n) dst[idx] = (_Float16)src[idx];
}

// f32 [R x C] -> f16 transposed [C x R]
__global__ void k_transpose_cast(const float* __restrict__ src, _Float16* __restrict__ dst,
                                 int R, int Cc) {
  int idx = blockIdx.x * blockDim.x + threadIdx.x;
  if (idx >= R * Cc) return;
  int r = idx / Cc, c = idx - r * Cc;
  dst[(long)c * R + r] = (_Float16)src[idx];
}

// ---------------------------------------------------------------------------
// Node embedding: x[node,o] = sum_{d<56} h[node,d] * W[o,d]   (K=56, VALU)
// ---------------------------------------------------------------------------
__global__ void k_embed(const float* __restrict__ h, const float* __restrict__ W,
                        float* __restrict__ x, _Float16* __restrict__ x16, int nodes) {
  int idx = blockIdx.x * blockDim.x + threadIdx.x;
  if (idx >= nodes * DD) return;
  int node = idx >> 7, o = idx & 127;
  const float* hr = h + (long)node * 56;
  const float* wr = W + (long)o * 56;
  float s = 0.0f;
#pragma unroll 8
  for (int d = 0; d < 56; ++d) s += hr[d] * wr[d];
  x[idx] = s;
  x16[idx] = (_Float16)s;
}

// ---------------------------------------------------------------------------
// GAT attention: att[b,i,:] = softmax_j( adj>0 ? e[b,i,j]+e[b,j,i] : -9e15 )*adj
// one block per (b,i); writes f32 att (unused downstream) + f16 att16 (GEMM A)
// ---------------------------------------------------------------------------
__global__ void k_att(const float* __restrict__ e, const float* __restrict__ adj,
                      float* __restrict__ att, _Float16* __restrict__ att16, int N) {
  const int i = blockIdx.x, b = blockIdx.y;
  __shared__ float sv[512];
  __shared__ float red[8];
  const float* erow = e + ((long)b * N + i) * N;
  const float* ecol = e + (long)b * N * N + i;
  const float* arow = adj + ((long)b * N + i) * N;
  const int lane = threadIdx.x & 31, wid = threadIdx.x >> 5;
  const int nw = blockDim.x >> 5;

  float lmax = -3.0e38f;
  for (int j = threadIdx.x; j < N; j += blockDim.x) {
    float a = arow[j];
    float v = (a > 0.0f) ? (erow[j] + ecol[(long)j * N]) : -9.0e15f;
    sv[j] = v;
    lmax = fmaxf(lmax, v);
  }
  for (int off = 16; off > 0; off >>= 1) lmax = fmaxf(lmax, __shfl_xor(lmax, off, 32));
  if (lane == 0) red[wid] = lmax;
  __syncthreads();
  float bmax = -3.0e38f;
  for (int w = 0; w < nw; ++w) bmax = fmaxf(bmax, red[w]);

  float lsum = 0.0f;
  for (int j = threadIdx.x; j < N; j += blockDim.x) {
    float p = __expf(sv[j] - bmax);
    sv[j] = p;
    lsum += p;
  }
  for (int off = 16; off > 0; off >>= 1) lsum += __shfl_xor(lsum, off, 32);
  __syncthreads();                       // everyone done reading red
  if (lane == 0) red[wid] = lsum;
  __syncthreads();
  float bsum = 0.0f;
  for (int w = 0; w < nw; ++w) bsum += red[w];
  float inv = 1.0f / bsum;
  for (int j = threadIdx.x; j < N; j += blockDim.x) {
    float v = sv[j] * inv * arow[j];
    att[((long)b * N + i) * N + j] = v;
    att16[((long)b * N + i) * N + j] = (_Float16)v;
  }
}

// ---------------------------------------------------------------------------
// GAT gate: coeff = sigmoid([x,hp]@gw + gb);  xout = coeff*x + (1-coeff)*hp
// one wave per node (32 lanes x 4 channels)
// ---------------------------------------------------------------------------
__global__ void k_gate(const float* __restrict__ x, const float* __restrict__ hp,
                       const float* __restrict__ gw, const float* __restrict__ gb,
                       float* __restrict__ xout, _Float16* __restrict__ xo16, int total) {
  int node = blockIdx.x * (blockDim.x >> 5) + (threadIdx.x >> 5);
  int lane = threadIdx.x & 31;
  if (node >= total) return;
  const float* xr = x + (long)node * DD;
  const float* hr = hp + (long)node * DD;
  float s = 0.0f;
#pragma unroll
  for (int q = 0; q < 4; ++q) {
    int c = lane * 4 + q;
    s += xr[c] * gw[c] + hr[c] * gw[DD + c];
  }
  for (int off = 16; off > 0; off >>= 1) s += __shfl_xor(s, off, 32);
  float coeff = 1.0f / (1.0f + __expf(-(s + gb[0])));
#pragma unroll
  for (int q = 0; q < 4; ++q) {
    int c = lane * 4 + q;
    float v = coeff * xr[c] + (1.0f - coeff) * hr[c];
    xout[(long)node * DD + c] = v;
    xo16[(long)node * DD + c] = (_Float16)v;
  }
}

// ---------------------------------------------------------------------------
// Output init: zero all 52 outputs, set hydro / flex / intercept columns
// ---------------------------------------------------------------------------
__global__ void k_init(const float* __restrict__ sasa, const float* __restrict__ rotor,
                       const float* __restrict__ sc, const float* __restrict__ rc,
                       const float* __restrict__ inter, float* __restrict__ out) {
  int t = threadIdx.x;
  if (t < 52) out[t] = 0.0f;
  __syncthreads();
  if (t < BB) {
    out[t * 13 + 8]  = fminf(sc[0] * sasa[t], 0.0f);
    out[t * 13 + 9]  = fmaxf(rc[0] * rotor[t], 0.0f);
    out[t * 13 + 12] = inter[0];
  }
}

// ---------------------------------------------------------------------------
// Sparse pairwise interaction energies (A_int gating).
// grid(i=64, b=4), block 256 = 8 waves; wave handles j = wid, wid+8, ...
// All 32 lanes cooperate per (pair,t): lane owns hidden channels lane+32q.
// ---------------------------------------------------------------------------
__global__ void k_pairs(const float* __restrict__ PA1, const float* __restrict__ PA2,
                        const float* __restrict__ PB1, const float* __restrict__ PB2,
                        const float* __restrict__ b1A, const float* __restrict__ b2A,
                        const float* __restrict__ w2A,
                        const float* __restrict__ b1B, const float* __restrict__ b2B,
                        const float* __restrict__ w2B,
                        const float* __restrict__ Aint, const float* __restrict__ dmv,
                        const float* __restrict__ Cparam, float* __restrict__ out) {
  const float SIG[NI] = {1.159f, 0.448f, 0.927f, 0.902f, 0.349f, 0.789f, 0.198f, 0.317f};
  const int i = blockIdx.x, b = blockIdx.y;
  const int wid = threadIdx.x >> 5, lane = threadIdx.x & 31;
  float acc[NI];
#pragma unroll
  for (int t = 0; t < NI; ++t) acc[t] = 0.0f;

  for (int t = 0; t < NI; ++t) {
    float p1a[4], p1b[4], wa[4], wb[4];
    const float* r1a = PA1 + (((long)t * BB + b) * N1 + i) * DD;
    const float* r1b = PB1 + (((long)t * BB + b) * N1 + i) * DD;
#pragma unroll
    for (int q = 0; q < 4; ++q) {
      int c = lane + 32 * q;
      p1a[q] = r1a[c] + b1A[t * DD + c];
      p1b[q] = r1b[c] + b1B[t * DD + c];
      wa[q]  = w2A[t * DD + c];
      wb[q]  = w2B[t * DD + c];
    }
    const float invs = 1.0f / (4.0f * SIG[t] * SIG[t]);
    const float Ct = Cparam[t];
    const float ba2 = b2A[t], bb2 = b2B[t];
    const float* P2At = PA2 + ((long)t * BB + b) * N2 * DD;
    const float* P2Bt = PB2 + ((long)t * BB + b) * N2 * DD;
    const float* AiRow = Aint + (((long)b * NI + t) * N1 + i) * N2;

    for (int j = wid; j < N2; j += 8) {
      float aint = AiRow[j];          // uniform across the wave
      if (aint == 0.0f) continue;     // ~98% sparse: skip both MLPs
      const float* dp = dmv + (((long)b * N1 + i) * N2 + j) * 3;
      float dx = dp[0], dy = dp[1], dz = dp[2];
      float dm = sqrtf(dx * dx + dy * dy + dz * dz + 1e-10f);
      if (dm < 0.5f) dm = 1e10f;
      const float* ra = P2At + (long)j * DD;
      const float* rb = P2Bt + (long)j * DD;
      float sA = 0.0f, sB = 0.0f;
#pragma unroll
      for (int q = 0; q < 4; ++q) {
        int c = lane + 32 * q;
        float hA = fmaxf(p1a[q] + ra[c], 0.0f);
        float hB = fmaxf(p1b[q] + rb[c], 0.0f);
        sA += hA * wa[q];
        sB += hB * wb[q];
      }
      for (int off = 16; off > 0; off >>= 1) {
        sA += __shfl_xor(sA, off, 32);
        sB += __shfl_xor(sB, off, 32);
      }
      float Ai = 4.0f / (1.0f + __expf(-(sA + ba2)));
      if (t == 1) Ai += 1.0f;
      float Bi = (3.0f / (1.0f + __expf(-(sB + bb2))) + 1.0f) * invs;
      float d = dm - Ct;
      float ev = Ai * (Bi * d * d - 1.0f) * aint;
      acc[t] += fminf(ev, 0.0f);
    }
  }
  if (lane == 0) {
#pragma unroll
    for (int t = 0; t < NI; ++t) atomicAdd(&out[b * 13 + t], acc[t]);
  }
}

// ---------------------------------------------------------------------------
// Coulomb + VdW over all pairs; wave-reduce then one atomicAdd per wave
// ---------------------------------------------------------------------------
__global__ void k_physics(const float* __restrict__ dmv,
                          const float* __restrict__ q1, const float* __restrict__ q2,
                          const float* __restrict__ r1, const float* __restrict__ r2,
                          const float* __restrict__ v1, const float* __restrict__ v2,
                          const float* __restrict__ nm1, const float* __restrict__ nm2,
                          const float* __restrict__ cc, const float* __restrict__ vc,
                          float* __restrict__ out) {
  const int b = blockIdx.y;
  const int pair = blockIdx.x * blockDim.x + threadIdx.x;   // < N1*N2
  const int i = pair >> 9, j = pair & (N2 - 1);
  const float* dp = dmv + (((long)b * N1 + i) * N2 + j) * 3;
  float dx = dp[0], dy = dp[1], dz = dp[2];
  float dm = sqrtf(dx * dx + dy * dy + dz * dz + 1e-10f);
  if (dm < 0.5f) dm = 1e10f;
  float coul = cc[0] * q1[b * N1 + i] * q2[b * N2 + j] / dm;
  coul *= v1[b * N1 + i] * v2[b * N2 + j];
  coul = fminf(fmaxf(coul, -100.0f), 100.0f);
  float rr = (r1[b * N1 + i] + r2[b * N2 + j]) / dm;
  float r2_ = rr * rr, r4 = r2_ * r2_, r8 = r4 * r4;
  float vdw = vc[0] * (r8 - 2.0f * r4) * nm1[b * N1 + i] * nm2[b * N2 + j];
  vdw = fminf(vdw, 100.0f);
  for (int off = 16; off > 0; off >>= 1) {
    coul += __shfl_xor(coul, off, 32);
    vdw  += __shfl_xor(vdw, off, 32);
  }
  if ((threadIdx.x & 31) == 0) {
    atomicAdd(&out[b * 13 + 10], coul);
    atomicAdd(&out[b * 13 + 11], vdw);
  }
}

// ---------------------------------------------------------------------------
// Host-side GAT layer: 4 all-f16 WMMA GEMMs + softmax + gate
// ---------------------------------------------------------------------------
static void gat_layer(const float* xin, const _Float16* xin16,
                      float* xout, _Float16* xout16,
                      const float* adj, int N,
                      const _Float16* W16l, const float* Wbl, const _Float16* A16Tl,
                      const float* gwl, const float* gbl,
                      float* H, _Float16* H16, _Float16* H16T, _Float16* HA16,
                      float* HP, float* E, float* ATT, _Float16* ATT16,
                      hipStream_t stream) {
  const int M = BB * N;
  // h = x @ W.T + Wb      -> f32 H (not read), f16 H16 (A/B), f16 H16T (hp's B)
  k_gemm_wmma<<<dim3(DD / 64, M / 16, 1), 32, 0, stream>>>(
      xin16, DD, 0L, W16l, DD, 0L, H, DD, 0L, H16, H16T, M, Wbl, 0, M, DD, DD);
  // hA = h @ A            (A pre-transposed to f16) -> only f16 HA16 needed
  k_gemm_wmma<<<dim3(DD / 64, M / 16, 1), 32, 0, stream>>>(
      H16, DD, 0L, A16Tl, DD, 0L, nullptr, DD, 0L, HA16, nullptr, 0,
      nullptr, 0, M, DD, DD);
  // e[b] = hA[b] @ h[b]^T (batched over b)
  k_gemm_wmma<<<dim3(N / 64, N / 16, BB), 32, 0, stream>>>(
      HA16, DD, (long)N * DD, H16, DD, (long)N * DD,
      E, N, (long)N * N, nullptr, nullptr, 0, nullptr, 0, N, N, DD);
  // att = softmax(mask(e + e^T)) * adj   -> f32 + f16
  k_att<<<dim3(N, BB), 256, 0, stream>>>(E, adj, ATT, ATT16, N);
  // hp = relu(att @ h)    (batched; B = h^T slice per batch, contiguous K)
  k_gemm_wmma<<<dim3(DD / 64, N / 16, BB), 32, 0, stream>>>(
      ATT16, N, (long)N * N, H16T + 0, M, (long)N,
      HP, DD, (long)N * DD, nullptr, nullptr, 0, nullptr, 1, N, DD, N);
  // gate
  k_gate<<<dim3((M + 7) / 8), 256, 0, stream>>>(xin, HP, gwl, gbl, xout, xout16, M);
}

extern "C" void kernel_launch(void* const* d_in, const int* in_sizes, int n_in,
                              void* d_out, int out_size, void* d_ws, size_t ws_size,
                              hipStream_t stream) {
  const float* h1    = (const float*)d_in[0];
  const float* adj1  = (const float*)d_in[1];
  const float* h2    = (const float*)d_in[2];
  const float* adj2  = (const float*)d_in[3];
  const float* Aint  = (const float*)d_in[4];
  const float* dmv   = (const float*)d_in[5];
  const float* sasa  = (const float*)d_in[6];
  const float* rotor = (const float*)d_in[7];
  const float* q1    = (const float*)d_in[8];
  const float* q2    = (const float*)d_in[9];
  const float* vr1   = (const float*)d_in[10];
  const float* vr2   = (const float*)d_in[11];
  const float* val1  = (const float*)d_in[12];
  const float* val2  = (const float*)d_in[13];
  const float* nm1   = (const float*)d_in[14];
  const float* nm2   = (const float*)d_in[15];
  const float* nodeW = (const float*)d_in[16];
  const float* gW    = (const float*)d_in[17];
  const float* gWb   = (const float*)d_in[18];
  const float* gA    = (const float*)d_in[19];
  const float* ggw   = (const float*)d_in[20];
  const float* ggb   = (const float*)d_in[21];
  const float* cAw1  = (const float*)d_in[22];
  const float* cAb1  = (const float*)d_in[23];
  const float* cAw2  = (const float*)d_in[24];
  const float* cAb2  = (const float*)d_in[25];
  const float* cBw1  = (const float*)d_in[26];
  const float* cBb1  = (const float*)d_in[27];
  const float* cBw2  = (const float*)d_in[28];
  const float* cBb2  = (const float*)d_in[29];
  const float* Cpar  = (const float*)d_in[30];
  const float* scff  = (const float*)d_in[31];
  const float* rcff  = (const float*)d_in[32];
  const float* ccff  = (const float*)d_in[33];
  const float* vcff  = (const float*)d_in[34];
  const float* intc  = (const float*)d_in[35];
  float* out = (float*)d_out;

  // ---- f32 workspace arena (~33 MB) --------------------------------------
  float* ws  = (float*)d_ws;
  float* X1a = ws;                         // 4*64*128
  float* X1b = X1a + BB * N1 * DD;
  float* X2a = X1b + BB * N1 * DD;         // 4*512*128
  float* X2b = X2a + BB * N2 * DD;
  float* H   = X2b + BB * N2 * DD;         // sized for N2, reused for N1
  float* HP  = H  + BB * N2 * DD;
  float* E   = HP + BB * N2 * DD;          // 4*512*512
  float* ATT = E  + BB * N2 * N2;
  float* PA1 = ATT + BB * N2 * N2;         // 8*4*64*128
  float* PB1 = PA1 + NI * BB * N1 * DD;
  float* PA2 = PB1 + NI * BB * N1 * DD;    // 8*4*512*128
  float* PB2 = PA2 + NI * BB * N2 * DD;

  // ---- f16 workspace arena (~6 MB, 32B-aligned) ---------------------------
  _Float16* f16b   = (_Float16*)(PB2 + NI * BB * N2 * DD);
  _Float16* X1a16  = f16b;                    _Float16* X1b16 = X1a16 + BB * N1 * DD;
  _Float16* X2a16  = X1b16 + BB * N1 * DD;    _Float16* X2b16 = X2a16 + BB * N2 * DD;
  _Float16* H16    = X2b16 + BB * N2 * DD;
  _Float16* H16T   = H16   + BB * N2 * DD;
  _Float16* HA16   = H16T  + BB * N2 * DD;
  _Float16* ATT16  = HA16  + BB * N2 * DD;    // 4*512*512
  _Float16* gW16   = ATT16 + BB * N2 * N2;    // 3*128*128
  _Float16* gA16T  = gW16  + LL * DD * DD;
  _Float16* w1A16  = gA16T + LL * DD * DD;    // 8*128*256
  _Float16* w1B16  = w1A16 + NI * DD * 2 * DD;

  // 0) weight pre-pass: cast (and transpose gnn_A) to f16
  k_cast<<<(LL * DD * DD + 255) / 256, 256, 0, stream>>>(gW, gW16, LL * DD * DD);
  k_cast<<<(NI * DD * 2 * DD + 255) / 256, 256, 0, stream>>>(cAw1, w1A16, NI * DD * 2 * DD);
  k_cast<<<(NI * DD * 2 * DD + 255) / 256, 256, 0, stream>>>(cBw1, w1B16, NI * DD * 2 * DD);
  for (int l = 0; l < LL; ++l)
    k_transpose_cast<<<(DD * DD + 255) / 256, 256, 0, stream>>>(
        gA + (long)l * DD * DD, gA16T + (long)l * DD * DD, DD, DD);

  // 1) node embedding (K=56 -> VALU), f32 + f16 twins
  k_embed<<<(BB * N1 * DD + 255) / 256, 256, 0, stream>>>(h1, nodeW, X1a, X1a16, BB * N1);
  k_embed<<<(BB * N2 * DD + 255) / 256, 256, 0, stream>>>(h2, nodeW, X2a, X2a16, BB * N2);

  // 2) three GAT layers on each graph (shared weights per layer)
  float* x1c = X1a; float* x1n = X1b;  _Float16* x1c16 = X1a16; _Float16* x1n16 = X1b16;
  float* x2c = X2a; float* x2n = X2b;  _Float16* x2c16 = X2a16; _Float16* x2n16 = X2b16;
  for (int l = 0; l < LL; ++l) {
    const _Float16* W16l  = gW16  + (long)l * DD * DD;
    const _Float16* A16Tl = gA16T + (long)l * DD * DD;
    const float* Wbl = gWb + (long)l * DD;
    const float* gwl = ggw + (long)l * 2 * DD;
    const float* gbl = ggb + l;
    gat_layer(x1c, x1c16, x1n, x1n16, adj1, N1, W16l, Wbl, A16Tl, gwl, gbl,
              H, H16, H16T, HA16, HP, E, ATT, ATT16, stream);
    gat_layer(x2c, x2c16, x2n, x2n16, adj2, N2, W16l, Wbl, A16Tl, gwl, gbl,
              H, H16, H16T, HA16, HP, E, ATT, ATT16, stream);
    float* t;    t = x1c; x1c = x1n; x1n = t;   t = x2c; x2c = x2n; x2n = t;
    _Float16* s; s = x1c16; x1c16 = x1n16; x1n16 = s;
                 s = x2c16; x2c16 = x2n16; x2n16 = s;
  }
  // final features now in x1c16 / x2c16

  // 3) separable layer-1 precompute: P = x @ w1_half.T  (all-f16 WMMA GEMMs)
  for (int t = 0; t < NI; ++t) {
    const _Float16* wA = w1A16 + (long)t * DD * 2 * DD;   // [128 hidden, 256 in]
    const _Float16* wB = w1B16 + (long)t * DD * 2 * DD;
    // graph1 (left half of w1), M = 256
    k_gemm_wmma<<<dim3(DD / 64, (BB * N1) / 16, 1), 32, 0, stream>>>(
        x1c16, DD, 0L, wA, 2 * DD, 0L, PA1 + (long)t * BB * N1 * DD, DD, 0L,
        nullptr, nullptr, 0, nullptr, 0, BB * N1, DD, DD);
    k_gemm_wmma<<<dim3(DD / 64, (BB * N1) / 16, 1), 32, 0, stream>>>(
        x1c16, DD, 0L, wB, 2 * DD, 0L, PB1 + (long)t * BB * N1 * DD, DD, 0L,
        nullptr, nullptr, 0, nullptr, 0, BB * N1, DD, DD);
    // graph2 (right half of w1), M = 2048
    k_gemm_wmma<<<dim3(DD / 64, (BB * N2) / 16, 1), 32, 0, stream>>>(
        x2c16, DD, 0L, wA + DD, 2 * DD, 0L, PA2 + (long)t * BB * N2 * DD, DD, 0L,
        nullptr, nullptr, 0, nullptr, 0, BB * N2, DD, DD);
    k_gemm_wmma<<<dim3(DD / 64, (BB * N2) / 16, 1), 32, 0, stream>>>(
        x2c16, DD, 0L, wB + DD, 2 * DD, 0L, PB2 + (long)t * BB * N2 * DD, DD, 0L,
        nullptr, nullptr, 0, nullptr, 0, BB * N2, DD, DD);
  }

  // 4) output init (zeros + hydro/flex/intercept), then sparse pair energies
  k_init<<<1, 64, 0, stream>>>(sasa, rotor, scff, rcff, intc, out);
  k_pairs<<<dim3(N1, BB), 256, 0, stream>>>(PA1, PA2, PB1, PB2,
                                            cAb1, cAb2, cAw2,
                                            cBb1, cBb2, cBw2,
                                            Aint, dmv, Cpar, out);
  k_physics<<<dim3((N1 * N2) / 256, BB), 256, 0, stream>>>(
      dmv, q1, q2, vr1, vr2, val1, val2, nm1, nm2, ccff, vcff, out);
}
